// WOP_36361193128453
// MI455X (gfx1250) — matrix operand order repaired and checked by reading, not compiled, not run on hardware
//
#include <hip/hip_runtime.h>
#include <hip/hip_bf16.h>
#include <stdint.h>

// ---------------------------------------------------------------------------
// MI455X (gfx1250): 2x 2-layer BiLSTM + attention head.
// Heavy GEMMs on v_wmma_f32_16x16x32_bf16; all operands padded so fragment
// loads are contiguous b128s; Kpad templated so B-tile addresses fold into
// immediate offsets (no spills).
// ---------------------------------------------------------------------------

typedef __bf16 v16bf __attribute__((ext_vector_type(16)));
typedef __bf16 v8bf  __attribute__((ext_vector_type(8)));
typedef float  v8f   __attribute__((ext_vector_type(8)));
typedef int    v4i   __attribute__((vector_size(16)));

#define DEVI static __device__ __forceinline__

DEVI __bf16 f2bf(float f) {
    unsigned u = __float_as_uint(f);
    u += 0x7fffu + ((u >> 16) & 1u);          // round-to-nearest-even
    unsigned short h = (unsigned short)(u >> 16);
    union { unsigned short s; __bf16 b; } cv; cv.s = h;
    return cv.b;
}

DEVI float sigm(float x) { return 1.0f / (1.0f + __expf(-x)); }

// CDNA5 async global->LDS path (ASYNCcnt), with fallback.
#if defined(__has_builtin)
#if __has_builtin(__builtin_amdgcn_global_load_async_to_lds_b128) && \
    __has_builtin(__builtin_amdgcn_s_wait_asynccnt)
#define HAVE_ASYNC_LDS 1
#endif
#endif

#ifdef HAVE_ASYNC_LDS
DEVI void async_copy_b128(const void* g, void* l) {
    __builtin_amdgcn_global_load_async_to_lds_b128(
        (__attribute__((address_space(1))) v4i*)(uintptr_t)g,
        (__attribute__((address_space(3))) v4i*)(l),
        0, 0);
}
#endif

// ---------------------------------------------------------------------------
// pad + convert fp32 -> bf16:  dst[dstRows, dstK], zero outside [srcRows, srcK]
// ---------------------------------------------------------------------------
__global__ void k_pad2d_bf16(const float* __restrict__ src, __bf16* __restrict__ dst,
                             int srcRows, int srcK, long dstRows, int dstK) {
    long i = (long)blockIdx.x * blockDim.x + threadIdx.x;
    if (i >= dstRows * dstK) return;
    long r = i / dstK; int k = (int)(i % dstK);
    float v = (r < srcRows && k < srcK) ? src[r * (long)srcK + k] : 0.0f;
    dst[i] = f2bf(v);
}

// bias = bih + bhh, padded to 208 with zeros
__global__ void k_bias_combine(const float* __restrict__ a, const float* __restrict__ b,
                               float* __restrict__ c) {
    int i = threadIdx.x;             // 256 threads, 1 block
    if (i < 208) c[i] = (i < 200) ? (a[i] + b[i]) : 0.0f;
}

// ---------------------------------------------------------------------------
// C[M,200] = A[M,KP](bf16) @ W[208,KP](bf16)^T + bias
// One wave per 16-row stripe; 13 accumulators (N = 13 x 16 = 208, store 200).
// KP compile-time: B-tile addresses = one base + immediate offsets.
// Fragment loads are contiguous:
//   A 16x32: lanes 0-15 K=0..15, lanes 16-31 K=16..31
//   B 32x16: lane = N, lanes 0-15 K=0..15, lanes 16-31 K=16..31
// ---------------------------------------------------------------------------
template <int KP>
__global__ __launch_bounds__(32, 1) void k_gemm_bf16(const __bf16* __restrict__ A,
                                                     const __bf16* __restrict__ W,
                                                     const float* __restrict__ bias,
                                                     float* __restrict__ C) {
    const int lane = threadIdx.x;             // 32 = one wave
    const bool hi  = lane >= 16;
    const int l15  = lane & 15;
    const long mb  = (long)blockIdx.x * 16;

    v8f acc[13] = {};
    const __bf16* arow = A + (mb + l15) * KP;
    const __bf16* wrow = W + l15 * KP;        // + t*16*KP is an immediate

    for (int kk = 0; kk < KP; kk += 32) {
        const int abase = kk + (hi ? 16 : 0);
        v8bf alo = *reinterpret_cast<const v8bf*>(arow + abase);
        v8bf ahi = *reinterpret_cast<const v8bf*>(arow + abase + 8);
        v16bf a = __builtin_shufflevector(alo, ahi, 0, 1, 2, 3, 4, 5, 6, 7,
                                          8, 9, 10, 11, 12, 13, 14, 15);
        const int bbase = kk + (hi ? 16 : 0);
#pragma unroll
        for (int t = 0; t < 13; ++t) {
            v16bf b = *reinterpret_cast<const v16bf*>(wrow + t * 16 * KP + bbase);
            acc[t] = __builtin_amdgcn_wmma_f32_16x16x32_bf16(false, a, false, b, (short)0,
                                                             acc[t], false, false);
        }
    }

    const long rb = mb + (hi ? 8 : 0);
#pragma unroll
    for (int t = 0; t < 13; ++t) {
        int col = t * 16 + l15;
        if (col < 200) {
            float bv = bias[col];
#pragma unroll
            for (int r = 0; r < 8; ++r) C[(rb + r) * 200 + col] = acc[t][r] + bv;
        }
    }
}

// ---------------------------------------------------------------------------
// Masked LSTM scan, 16 sequences per block (B is always a multiple of 16).
// pre[B,T,200] = x@Wih^T + (bih+bhh).  Per step: gates += h @ Whh^T via WMMA
// (M=16 seqs, N=13x16, K=50 padded to 64).  whh is padded [208,64] bf16 and
// staged in LDS.  rev selects time direction; updates masked by t < len.
// ---------------------------------------------------------------------------
__global__ __launch_bounds__(128, 1) void k_lstm_scan(const float* __restrict__ pre,
                                                      const __bf16* __restrict__ whh,
                                                      const int* __restrict__ lens,
                                                      float* __restrict__ out,
                                                      float* __restrict__ hT,
                                                      int B, int T, int rev) {
    __shared__ __bf16 sh_h[16][64];       // h state (bf16), cols >= 50 stay 0
    __shared__ __bf16 sh_w[208 * 64];     // padded Whh
    __shared__ float  sh_c[16][50];
    __shared__ float  sh_g[16][208];      // gates (f32), cols >= 200 stay 0
    __shared__ int    sh_len[16];

    const int tid  = threadIdx.x;         // 128 = 4 waves
    const int wv   = tid >> 5;
    const int lane = tid & 31;
    const bool hi  = lane >= 16;
    const int l15  = lane & 15;
    const int sb   = blockIdx.x * 16;

    // init state + stage weights (vectorized: 16B chunks)
    for (int i = tid; i < 16 * 64 / 8; i += 128)
        reinterpret_cast<v8bf*>(&sh_h[0][0])[i] = v8bf{};
    for (int i = tid; i < 16 * 50; i += 128) sh_c[i / 50][i % 50] = 0.0f;
    for (int i = tid; i < 16 * 208; i += 128) sh_g[i / 208][i % 208] = 0.0f;
    for (int i = tid; i < 208 * 64 / 8; i += 128)
        reinterpret_cast<v8bf*>(sh_w)[i] = reinterpret_cast<const v8bf*>(whh)[i];
    if (tid < 16) sh_len[tid] = lens[sb + tid];
    __syncthreads();

    for (int step = 0; step < T; ++step) {
        const int t = rev ? (T - 1 - step) : step;

        // phase 1: stage pre[seq, t, 0:200] -> sh_g   (800 x 16B chunks)
        for (int i = tid; i < 800; i += 128) {
            int s = i / 50, cj = i % 50;
            const float* g = pre + ((size_t)(sb + s) * T + t) * 200 + cj * 4;
#ifdef HAVE_ASYNC_LDS
            async_copy_b128(g, &sh_g[s][cj * 4]);
#else
            *reinterpret_cast<float4*>(&sh_g[s][cj * 4]) =
                *reinterpret_cast<const float4*>(g);
#endif
        }
#ifdef HAVE_ASYNC_LDS
        __builtin_amdgcn_s_wait_asynccnt(0);
#endif
        __syncthreads();

        // phase 2: gates += h @ Whh^T   (waves own N tiles wv, wv+4, ...)
        for (int tile = wv; tile < 13; tile += 4) {
            const int col = tile * 16 + l15;
            const int rb  = hi ? 8 : 0;
            v8f acc;
#pragma unroll
            for (int r = 0; r < 8; ++r) acc[r] = sh_g[rb + r][col];
#pragma unroll
            for (int ks = 0; ks < 2; ++ks) {
                const int kk = ks * 32;
                v8bf alo = *reinterpret_cast<const v8bf*>(&sh_h[l15][kk + (hi ? 16 : 0)]);
                v8bf ahi = *reinterpret_cast<const v8bf*>(&sh_h[l15][kk + (hi ? 16 : 0) + 8]);
                v16bf a = __builtin_shufflevector(alo, ahi, 0, 1, 2, 3, 4, 5, 6, 7,
                                                  8, 9, 10, 11, 12, 13, 14, 15);
                v16bf b = *reinterpret_cast<const v16bf*>(&sh_w[col * 64 + kk + (hi ? 16 : 0)]);
                acc = __builtin_amdgcn_wmma_f32_16x16x32_bf16(false, a, false, b, (short)0,
                                                              acc, false, false);
            }
#pragma unroll
            for (int r = 0; r < 8; ++r) sh_g[rb + r][col] = acc[r];
        }
        __syncthreads();

        // phase 3: activations + masked update (gate order i,f,g,o)
        for (int i = tid; i < 16 * 50; i += 128) {
            int s = i / 50, u = i % 50;
            float gi = sh_g[s][u];
            float gf = sh_g[s][50 + u];
            float gg = sh_g[s][100 + u];
            float go = sh_g[s][150 + u];
            float cn = sigm(gf) * sh_c[s][u] + sigm(gi) * tanhf(gg);
            float hn = sigm(go) * tanhf(cn);
            bool act = (t < sh_len[s]);
            if (act) { sh_c[s][u] = cn; sh_h[s][u] = f2bf(hn); }
            out[((size_t)(sb + s) * T + t) * 50 + u] = act ? hn : 0.0f;
        }
        __syncthreads();
    }

    for (int i = tid; i < 16 * 50; i += 128) {
        int s = i / 50, u = i % 50;
        float hv = __bfloat162float(*(const __hip_bfloat16*)&sh_h[s][u]);
        hT[(size_t)(sb + s) * 50 + u] = hv;
    }
}

// ---------------------------------------------------------------------------
// layer-1 input: x1[row, 0:50]=out_f, [50:100]=out_b, [100:128]=0  (bf16)
// ---------------------------------------------------------------------------
__global__ void k_concat2_bf16(const float* __restrict__ f, const float* __restrict__ b,
                               __bf16* __restrict__ dst, long rows) {
    long i = (long)blockIdx.x * blockDim.x + threadIdx.x;
    if (i >= rows * 128) return;
    long r = i / 128; int c = (int)(i % 128);
    float v = (c < 50) ? f[r * 50 + c] : (c < 100 ? b[r * 50 + (c - 50)] : 0.0f);
    dst[i] = f2bf(v);
}

// wenc_n_ext[row,112] = [out_f1(50) | out_b1(50) | one_hot12(knowledge)]
__global__ void k_build_wenc_n(const float* __restrict__ f, const float* __restrict__ b,
                               const int* __restrict__ kn, float* __restrict__ dst, long rows) {
    long i = (long)blockIdx.x * blockDim.x + threadIdx.x;
    if (i >= rows * 112) return;
    long r = i / 112; int c = (int)(i % 112);
    float v;
    if (c < 50)       v = f[r * 50 + c];
    else if (c < 100) v = b[r * 50 + (c - 50)];
    else              v = (kn[r] == (c - 100)) ? 1.0f : 0.0f;
    dst[i] = v;
}

// wenc_hs_ext[row,104] = [hT_f1(50) | hT_b1(50) | one_hot4(knowledge_header)]
__global__ void k_build_wenc_hs(const float* __restrict__ f, const float* __restrict__ b,
                                const int* __restrict__ kh, float* __restrict__ dst, long rows) {
    long i = (long)blockIdx.x * blockDim.x + threadIdx.x;
    if (i >= rows * 104) return;
    long r = i / 104; int c = (int)(i % 104);
    float v;
    if (c < 50)       v = f[r * 50 + c];
    else if (c < 100) v = b[r * 50 + (c - 50)];
    else              v = (kh[r] == (c - 100)) ? 1.0f : 0.0f;
    dst[i] = v;
}

// ---------------------------------------------------------------------------
// Attention + output MLP, one block per batch element (256 threads).
//   att[w,n] = wenc[n,:112] . (hs_ob[w,:] @ W_att_w) + hs_ob[w,:] . W_att_b
// ---------------------------------------------------------------------------
__global__ void k_attn(const float* __restrict__ wenc, const float* __restrict__ hs,
                       const int* __restrict__ l_n, const int* __restrict__ wn,
                       const int* __restrict__ wc,
                       const float* __restrict__ Waw, const float* __restrict__ Wab,
                       const float* __restrict__ Wcw, const float* __restrict__ Wcb,
                       const float* __restrict__ Whsw, const float* __restrict__ Whsb,
                       const float* __restrict__ w1w, const float* __restrict__ w1b,
                       const float* __restrict__ w2w, const float* __restrict__ w2b,
                       float* __restrict__ out) {
    __shared__ float s_we[64 * 112];
    __shared__ float s_hs[4 * 104];
    __shared__ float s_hsW[4 * 112];
    __shared__ float s_hsb[4];
    __shared__ float s_att[4 * 64];
    __shared__ float s_cn[4 * 112];
    __shared__ float s_vec[4 * 200];
    __shared__ float s_s1[4 * 100];

    const int b = blockIdx.x, tid = threadIdx.x;
    const float* wb = wenc + (size_t)b * 64 * 112;

    for (int i = tid; i < 64 * 112; i += 256) s_we[i] = wb[i];
    const int wnb = wn[b];
    for (int i = tid; i < 4 * 104; i += 256) {
        int w = i / 104, d = i % 104;
        int idx = (w < wnb) ? wc[b * 4 + w] : 0;
        s_hs[i] = hs[((size_t)b * 16 + idx) * 104 + d];
    }
    __syncthreads();

    for (int i = tid; i < 4 * 112; i += 256) {
        int w = i / 112, d1 = i % 112;
        float acc = 0.0f;
        for (int d2 = 0; d2 < 104; ++d2) acc += s_hs[w * 104 + d2] * Waw[d2 * 112 + d1];
        s_hsW[i] = acc;
    }
    if (tid < 4) {
        float acc = 0.0f;
        for (int d2 = 0; d2 < 104; ++d2) acc += s_hs[tid * 104 + d2] * Wab[d2];
        s_hsb[tid] = acc;
    }
    __syncthreads();

    const int ln = l_n[b];
    for (int i = tid; i < 4 * 64; i += 256) {
        int w = i / 64, n = i % 64;
        float acc = s_hsb[w];
        for (int d = 0; d < 112; ++d) acc += s_we[n * 112 + d] * s_hsW[w * 112 + d];
        s_att[i] = (n < ln) ? acc : -1e10f;
    }
    __syncthreads();

    if (tid < 4) {   // masked softmax over n
        float mx = -3.0e38f;
        for (int n = 0; n < 64; ++n) mx = fmaxf(mx, s_att[tid * 64 + n]);
        float sum = 0.0f;
        for (int n = 0; n < 64; ++n) {
            float e = __expf(s_att[tid * 64 + n] - mx);
            s_att[tid * 64 + n] = e; sum += e;
        }
        float inv = 1.0f / sum;
        for (int n = 0; n < 64; ++n) s_att[tid * 64 + n] *= inv;
    }
    __syncthreads();

    for (int i = tid; i < 4 * 112; i += 256) {     // c_n = p @ wenc
        int w = i / 112, d = i % 112;
        float acc = 0.0f;
        for (int n = 0; n < 64; ++n) acc += s_att[w * 64 + n] * s_we[n * 112 + d];
        s_cn[i] = acc;
    }
    __syncthreads();

    for (int i = tid; i < 4 * 200; i += 256) {     // vec = [c_n@Wc^T | hs@Whs^T]
        int w = i / 200, d = i % 200;
        float acc;
        if (d < 100) {
            acc = Wcb[d];
            for (int k = 0; k < 112; ++k) acc += s_cn[w * 112 + k] * Wcw[d * 112 + k];
        } else {
            int dd = d - 100;
            acc = Whsb[dd];
            for (int k = 0; k < 104; ++k) acc += s_hs[w * 104 + k] * Whsw[dd * 104 + k];
        }
        s_vec[i] = acc;
    }
    __syncthreads();

    for (int i = tid; i < 4 * 100; i += 256) {     // tanh(vec @ wo1^T + b)
        int w = i / 100, d = i % 100;
        float acc = w1b[d];
        for (int k = 0; k < 200; ++k) acc += s_vec[w * 200 + k] * w1w[d * 200 + k];
        s_s1[i] = tanhf(acc);
    }
    __syncthreads();

    for (int i = tid; i < 12; i += 256) {          // @ wo2^T + b
        int w = i / 3, d = i % 3;
        float acc = w2b[d];
        for (int k = 0; k < 100; ++k) acc += s_s1[w * 100 + k] * w2w[d * 100 + k];
        out[(size_t)b * 12 + i] = acc;
    }
}

// ---------------------------------------------------------------------------
// host launcher
// ---------------------------------------------------------------------------
extern "C" void kernel_launch(void* const* d_in, const int* in_sizes, int n_in,
                              void* d_out, int out_size, void* d_ws, size_t ws_size,
                              hipStream_t stream) {
    (void)in_sizes; (void)n_in; (void)out_size; (void)ws_size;

    const float* wemb_n   = (const float*)d_in[0];
    const int*   l_n      = (const int*)d_in[1];
    const float* wemb_hpu = (const float*)d_in[2];
    const int*   l_hpu    = (const int*)d_in[3];
    // d_in[4] = l_hs (always MLHS, unused)
    const int*   wn        = (const int*)d_in[5];
    const int*   wc        = (const int*)d_in[6];
    const int*   knowledge = (const int*)d_in[7];
    const int*   khdr      = (const int*)d_in[8];

    const long NBT = 512L * 64;    // 32768 question rows
    const long HBT = 8192L * 8;    // 65536 header rows
    const int  KP0 = 320;          // layer0 K padded (300 -> 320)
    const int  KP1 = 128;          // layer1 K padded (100 -> 128)

    char* base = (char*)d_ws; size_t off = 0;
    auto alloc = [&](size_t bytes) -> void* {
        void* p = base + off; off = (off + bytes + 255) & ~(size_t)255; return p;
    };
    auto cdiv = [](long a, long b) -> unsigned { return (unsigned)((a + b - 1) / b); };

    // ---- padded bf16 weights + combined biases ------------------------------
    // sets: s = enc*4 + layer*2 + dir ; enc0 = n (base 9), enc1 = h (base 25)
    __bf16* WihBf[8]; __bf16* WhhBf[8]; float* BiasC[8];
    const int encBase[2] = {9, 25};
    for (int e = 0; e < 2; ++e)
        for (int l = 0; l < 2; ++l)
            for (int d = 0; d < 2; ++d) {
                int s  = e * 4 + l * 2 + d;
                int bi = encBase[e] + (l * 2 + d) * 4;   // Wih, Whh, bih, bhh
                int Kin  = (l == 0) ? 300 : 100;
                int Kpad = (l == 0) ? KP0 : KP1;
                WihBf[s] = (__bf16*)alloc((size_t)208 * Kpad * 2);
                WhhBf[s] = (__bf16*)alloc((size_t)208 * 64 * 2);
                BiasC[s] = (float*)alloc(208 * 4);
                k_pad2d_bf16<<<cdiv(208L * Kpad, 256), 256, 0, stream>>>(
                    (const float*)d_in[bi], WihBf[s], 200, Kin, 208, Kpad);
                k_pad2d_bf16<<<cdiv(208L * 64, 256), 256, 0, stream>>>(
                    (const float*)d_in[bi + 1], WhhBf[s], 200, 50, 208, 64);
                k_bias_combine<<<1, 256, 0, stream>>>(
                    (const float*)d_in[bi + 2], (const float*)d_in[bi + 3], BiasC[s]);
            }

    // ---- activation scratch -------------------------------------------------
    __bf16* nX   = (__bf16*)alloc((size_t)NBT * KP0 * 2);
    __bf16* hX   = (__bf16*)alloc((size_t)HBT * KP0 * 2);
    float* preNf = (float*)alloc((size_t)NBT * 200 * 4);   // reused L0 & L1
    float* preNb = (float*)alloc((size_t)NBT * 200 * 4);
    float* preHf = (float*)alloc((size_t)HBT * 200 * 4);
    float* preHb = (float*)alloc((size_t)HBT * 200 * 4);
    float* oNf0  = (float*)alloc((size_t)NBT * 50 * 4);
    float* oNb0  = (float*)alloc((size_t)NBT * 50 * 4);
    float* oHf0  = (float*)alloc((size_t)HBT * 50 * 4);
    float* oHb0  = (float*)alloc((size_t)HBT * 50 * 4);
    __bf16* x1N  = (__bf16*)alloc((size_t)NBT * KP1 * 2);
    __bf16* x1H  = (__bf16*)alloc((size_t)HBT * KP1 * 2);
    float* oNf1  = (float*)alloc((size_t)NBT * 50 * 4);
    float* oNb1  = (float*)alloc((size_t)NBT * 50 * 4);
    float* oHf1  = (float*)alloc((size_t)HBT * 50 * 4);
    float* oHb1  = (float*)alloc((size_t)HBT * 50 * 4);
    float* hTd   = (float*)alloc(8192L * 50 * 4);          // dump for unused finals
    float* hTHf  = (float*)alloc(8192L * 50 * 4);
    float* hTHb  = (float*)alloc(8192L * 50 * 4);
    float* wencN = (float*)alloc((size_t)NBT * 112 * 4);
    float* wencH = (float*)alloc(8192L * 104 * 4);

    // ---- convert + pad inputs ----------------------------------------------
    k_pad2d_bf16<<<cdiv(NBT * KP0, 256), 256, 0, stream>>>(wemb_n, nX, (int)NBT, 300, NBT, KP0);
    k_pad2d_bf16<<<cdiv(HBT * KP0, 256), 256, 0, stream>>>(wemb_hpu, hX, (int)HBT, 300, HBT, KP0);

    auto gemm = [&](const __bf16* A, int s, float* C, long M, int Kpad) {
        if (Kpad == 320)
            k_gemm_bf16<320><<<dim3((unsigned)(M / 16)), 32, 0, stream>>>(A, WihBf[s], BiasC[s], C);
        else
            k_gemm_bf16<128><<<dim3((unsigned)(M / 16)), 32, 0, stream>>>(A, WihBf[s], BiasC[s], C);
    };
    auto scan = [&](const float* pre, int s, const int* lens, float* o, float* hT,
                    int B, int T, int rev) {
        k_lstm_scan<<<dim3(B / 16), 128, 0, stream>>>(pre, WhhBf[s], lens, o, hT, B, T, rev);
    };

    // ---- layer 0 ------------------------------------------------------------
    gemm(nX, 0, preNf, NBT, KP0);  gemm(nX, 1, preNb, NBT, KP0);
    gemm(hX, 4, preHf, HBT, KP0);  gemm(hX, 5, preHb, HBT, KP0);
    scan(preNf, 0, l_n,   oNf0, hTd, 512, 64, 0);
    scan(preNb, 1, l_n,   oNb0, hTd, 512, 64, 1);
    scan(preHf, 4, l_hpu, oHf0, hTd, 8192, 8, 0);
    scan(preHb, 5, l_hpu, oHb0, hTd, 8192, 8, 1);

    // ---- layer 1 ------------------------------------------------------------
    k_concat2_bf16<<<cdiv(NBT * KP1, 256), 256, 0, stream>>>(oNf0, oNb0, x1N, NBT);
    k_concat2_bf16<<<cdiv(HBT * KP1, 256), 256, 0, stream>>>(oHf0, oHb0, x1H, HBT);
    gemm(x1N, 2, preNf, NBT, KP1);  gemm(x1N, 3, preNb, NBT, KP1);
    gemm(x1H, 6, preHf, HBT, KP1);  gemm(x1H, 7, preHb, HBT, KP1);
    scan(preNf, 2, l_n,   oNf1, hTd,  512, 64, 0);
    scan(preNb, 3, l_n,   oNb1, hTd,  512, 64, 1);
    scan(preHf, 6, l_hpu, oHf1, hTHf, 8192, 8, 0);
    scan(preHb, 7, l_hpu, oHb1, hTHb, 8192, 8, 1);

    // ---- feature concats ----------------------------------------------------
    k_build_wenc_n<<<cdiv(NBT * 112, 256), 256, 0, stream>>>(oNf1, oNb1, knowledge, wencN, NBT);
    k_build_wenc_hs<<<cdiv(8192L * 104, 256), 256, 0, stream>>>(hTHf, hTHb, khdr, wencH, 8192);

    // ---- attention + output MLP --------------------------------------------
    k_attn<<<dim3(512), 256, 0, stream>>>(
        wencN, wencH, l_n, wn, wc,
        (const float*)d_in[41], (const float*)d_in[42],   // W_att
        (const float*)d_in[43], (const float*)d_in[44],   // W_c
        (const float*)d_in[45], (const float*)d_in[46],   // W_hs
        (const float*)d_in[47], (const float*)d_in[48],   // wo1
        (const float*)d_in[49], (const float*)d_in[50],   // wo2
        (float*)d_out);
}
